// CoolingPhaseDetector_18167711662379
// MI455X (gfx1250) — matrix operand order repaired and checked by reading, not compiled, not run on hardware
//
#include <hip/hip_runtime.h>
#include <hip/hip_bf16.h>

typedef __attribute__((ext_vector_type(16))) _Float16 v16h;
typedef __attribute__((ext_vector_type(8)))  _Float16 v8h;
typedef __attribute__((ext_vector_type(4)))  _Float16 v4h;
typedef __attribute__((ext_vector_type(8)))  float    v8f;

#define BZ 32
#define TZ 4096
#define CZ 256
#define NROWS (BZ*TZ)          // 131072 flattened rows

// ---------------- helpers ----------------
static __device__ __forceinline__ v16h cat16(v8h lo, v8h hi) {
    return __builtin_shufflevector(lo, hi, 0,1,2,3,4,5,6,7,8,9,10,11,12,13,14,15);
}
// B fragment (32x16 f16): lane%16 = N, 16 consecutive halves along K
static __device__ __forceinline__ v16h ldB(const _Float16* p) {
    v8h lo = *(const v8h*)p;
    v8h hi = *(const v8h*)(p + 8);
    return cat16(lo, hi);
}
// A fragment (16x32 f16) from LDS row: halves 0..7 at col, 8..15 at col+16
static __device__ __forceinline__ v16h ldA(const _Float16* p) {
    v8h lo = *(const v8h*)p;
    v8h hi = *(const v8h*)(p + 16);
    return cat16(lo, hi);
}

// ---------------- kernel 0: weight prep (fp32 -> f16, pad N to 16) ----------------
__global__ void cpd_prep(const float* __restrict__ W1, const float* __restrict__ W2,
                         const float* __restrict__ W3, const float* __restrict__ cw,
                         _Float16* __restrict__ W1h, _Float16* __restrict__ W2h,
                         _Float16* __restrict__ W3h, _Float16* __restrict__ CWh)
{
    int i = blockIdx.x * 256 + threadIdx.x;
    if (i < 32768) { W1h[i] = (_Float16)W1[(i & 255) * 128 + (i >> 8)]; return; }
    int j = i - 32768;
    if (j < 8192)  { W2h[j] = (_Float16)W2[(j & 127) * 64 + (j >> 7)]; return; }
    j -= 8192;
    if (j < 1024)  { int r = j & 63, c = j >> 6;
                     W3h[j] = (c < 4) ? (_Float16)W3[r*4 + c] : (_Float16)0.0f; return; }
    j -= 1024;
    if (j < 4096)  { int r = j & 255, c = j >> 8;
                     CWh[j] = (c < 3) ? (_Float16)cw[r*3 + c] : (_Float16)0.0f; return; }
}

// ---------------- kernel 1: fused MLP + conv-dots via WMMA ----------------
// Block = 128 threads (4 waves) covering 64 rows. Waves split the *N* dimension
// so each W1/W2 fragment is loaded from L2 exactly once per block and reused
// across all four 16-row M-tiles (A operands re-read from LDS, which is cheap).
#define AST  264   // 256 + 8 pad -> 528B rows: 16B aligned, bank-conflict free
#define H1ST 136   // 128 + 8
#define H2ST 72    // 64 + 8

__global__ void __launch_bounds__(128)
cpd_mlp_wmma(const float* __restrict__ gru,
             const _Float16* __restrict__ W1h, const _Float16* __restrict__ W2h,
             const _Float16* __restrict__ W3h, const _Float16* __restrict__ CWh,
             const float* __restrict__ b1, const float* __restrict__ b2,
             const float* __restrict__ b3,
             float* __restrict__ out_pp,
             float* __restrict__ d0p, float* __restrict__ d1p, float* __restrict__ d2p,
             int* __restrict__ domp, float* __restrict__ confp)
{
    __shared__ _Float16 Atile[64 * AST];   // 33792 B
    __shared__ _Float16 H1t [64 * H1ST];   // 17408 B
    __shared__ _Float16 H2t [64 * H2ST];   //  9216 B

    const int tid = threadIdx.x;
    const int blockRow0 = blockIdx.x * 64;

    // ---- stage 64x256 fp32 rows -> f16 LDS (coalesced float4 loads) ----
    const float4* gsrc = (const float4*)(gru + (size_t)blockRow0 * CZ);
    for (int i = tid; i < 64 * 64; i += 128) {
        int row = i >> 6, c4 = i & 63;
        float4 v = gsrc[(size_t)row * 64 + c4];
        v4h h = { (_Float16)v.x, (_Float16)v.y, (_Float16)v.z, (_Float16)v.w };
        *(v4h*)(Atile + row * AST + c4 * 4) = h;
    }
    __syncthreads();

    const int lane = tid & 31;
    const int wv   = tid >> 5;                // wave id 0..3
    const int m    = lane & 15;               // M for A-frag / N for D-frag
    const int grp  = lane >> 4;

    // ---- stage 1: h1 = relu(A @ W1 + b1); wave wv owns N-tiles {2wv, 2wv+1} ----
#pragma unroll
    for (int q = 0; q < 2; ++q) {
        const int nt = wv * 2 + q;
        const float bv = b1[nt * 16 + m];
        v8f acc[4];
#pragma unroll
        for (int mt = 0; mt < 4; ++mt)
            acc[mt] = (v8f){ bv, bv, bv, bv, bv, bv, bv, bv };
#pragma unroll
        for (int c = 0; c < 8; ++c) {
            v16h bf = ldA(W1h + (size_t)(nt * 16 + m) * 256 + c * 32 + grp * 8);
#pragma unroll
            for (int mt = 0; mt < 4; ++mt) {
                v16h a = ldA(Atile + (mt * 16 + m) * AST + c * 32 + grp * 8);
                acc[mt] = __builtin_amdgcn_wmma_f32_16x16x32_f16(false, a, false, bf,
                                                                 (short)0, acc[mt], false, false);
            }
        }
#pragma unroll
        for (int mt = 0; mt < 4; ++mt)
#pragma unroll
            for (int v = 0; v < 8; ++v) {
                float x = acc[mt][v]; x = x > 0.0f ? x : 0.0f;
                H1t[(mt * 16 + v + grp * 8) * H1ST + nt * 16 + m] = (_Float16)x;
            }
    }

    // ---- stage 1b: conv dots; wave wv owns M-tile wv ----
    {
        v8f acc = { 0.f, 0.f, 0.f, 0.f, 0.f, 0.f, 0.f, 0.f };
#pragma unroll
        for (int c = 0; c < 8; ++c) {
            v16h bf = ldA(CWh + (size_t)m * 256 + c * 32 + grp * 8);
            v16h a  = ldA(Atile + (wv * 16 + m) * AST + c * 32 + grp * 8);
            acc = __builtin_amdgcn_wmma_f32_16x16x32_f16(false, a, false, bf,
                                                         (short)0, acc, false, false);
        }
        const size_t r0 = (size_t)blockRow0 + wv * 16;
        if (m < 3) {
            float* dp = (m == 0) ? d0p : ((m == 1) ? d1p : d2p);
#pragma unroll
            for (int v = 0; v < 8; ++v)
                dp[r0 + v + grp * 8] = acc[v];
        }
    }
    __syncthreads();   // h1 complete across waves

    // ---- stage 2: h2 = relu(h1 @ W2 + b2); wave wv owns N-tile wv ----
    {
        const int nt = wv;
        const float bv = b2[nt * 16 + m];
        v8f acc[4];
#pragma unroll
        for (int mt = 0; mt < 4; ++mt)
            acc[mt] = (v8f){ bv, bv, bv, bv, bv, bv, bv, bv };
#pragma unroll
        for (int c = 0; c < 4; ++c) {
            v16h bf = ldA(W2h + (size_t)(nt * 16 + m) * 128 + c * 32 + grp * 8);
#pragma unroll
            for (int mt = 0; mt < 4; ++mt) {
                v16h a = ldA(H1t + (mt * 16 + m) * H1ST + c * 32 + grp * 8);
                acc[mt] = __builtin_amdgcn_wmma_f32_16x16x32_f16(false, a, false, bf,
                                                                 (short)0, acc[mt], false, false);
            }
        }
#pragma unroll
        for (int mt = 0; mt < 4; ++mt)
#pragma unroll
            for (int v = 0; v < 8; ++v) {
                float x = acc[mt][v]; x = x > 0.0f ? x : 0.0f;
                H2t[(mt * 16 + v + grp * 8) * H2ST + nt * 16 + m] = (_Float16)x;
            }
    }
    __syncthreads();   // h2 complete across waves

    // ---- stage 3: logits = h2 @ W3pad + b3; wave wv owns M-tile wv ----
    const size_t r0 = (size_t)blockRow0 + wv * 16;
    float bv3 = (m < 4) ? b3[m] : 0.0f;
    v8f acc3 = { bv3, bv3, bv3, bv3, bv3, bv3, bv3, bv3 };
#pragma unroll
    for (int c = 0; c < 2; ++c) {
        v16h bf = ldA(W3h + (size_t)m * 64 + c * 32 + grp * 8);
        v16h a  = ldA(H2t + (wv * 16 + m) * H2ST + c * 32 + grp * 8);
        acc3 = __builtin_amdgcn_wmma_f32_16x16x32_f16(false, a, false, bf,
                                                      (short)0, acc3, false, false);
    }

    // ---- softmax/argmax over 4 logits per row (cross-lane via shuffles) ----
    const int base = grp * 16;
#pragma unroll
    for (int v = 0; v < 8; ++v) {
        float x  = acc3[v];
        float l0 = __shfl(x, base + 0, 32);
        float l1 = __shfl(x, base + 1, 32);
        float l2 = __shfl(x, base + 2, 32);
        float l3 = __shfl(x, base + 3, 32);
        float mx = fmaxf(fmaxf(l0, l1), fmaxf(l2, l3));
        float e0 = __expf(l0 - mx), e1 = __expf(l1 - mx);
        float e2 = __expf(l2 - mx), e3 = __expf(l3 - mx);
        float inv = 1.0f / (e0 + e1 + e2 + e3);
        size_t gr = r0 + v + grp * 8;
        if (m < 4) {
            float pm = (m == 0 ? e0 : m == 1 ? e1 : m == 2 ? e2 : e3) * inv;
            out_pp[gr * 4 + m] = pm;
        }
        if (m == 0) {
            int best = 0; float bl = l0;
            if (l1 > bl) { bl = l1; best = 1; }
            if (l2 > bl) { bl = l2; best = 2; }
            if (l3 > bl) { bl = l3; best = 3; }
            float eb = (best == 0 ? e0 : best == 1 ? e1 : best == 2 ? e2 : e3);
            domp[gr]  = best;
            confp[gr] = eb * inv;
        }
    }
}

// ---------------- kernel 2: transition scores + change bits ----------------
__global__ void cpd_trans(const float* __restrict__ d0p, const float* __restrict__ d1p,
                          const float* __restrict__ d2p, const int* __restrict__ domp,
                          const float* __restrict__ cb,
                          float* __restrict__ out_ts, int* __restrict__ chg)
{
    int g = blockIdx.x * 256 + threadIdx.x;
    if (g >= NROWS) return;
    int t = g & (TZ - 1);
    float x = d1p[g] + cb[0];
    if (t > 0)       x += d0p[g - 1];
    if (t < TZ - 1)  x += d2p[g + 1];
    float ts = 1.0f / (1.0f + __expf(-x));
    out_ts[g] = ts;
    int c = 0;
    if (t > 0) c = ((domp[g] != domp[g - 1]) || (ts > 0.7f)) ? 1 : 0;
    chg[g] = c;
}

// ---------------- kernel 3: per-batch scan + contiguous-segment means ----------------
__global__ void __launch_bounds__(256)
cpd_seg(const int* __restrict__ chg, const float* __restrict__ confp,
        const float* __restrict__ inten,
        int* __restrict__ out_seg, float* __restrict__ out_mi, float* __restrict__ out_mc)
{
    __shared__ float segc[TZ];
    __shared__ float segi[TZ];
    __shared__ float segf[TZ];
    __shared__ int   sbuf[256];

    const int b = blockIdx.x, tid = threadIdx.x;
    const size_t base = (size_t)b * TZ + (size_t)tid * 16;

    for (int i = tid; i < TZ; i += 256) { segc[i] = 0.f; segi[i] = 0.f; segf[i] = 0.f; }

    int ls[16]; int run = 0;
#pragma unroll
    for (int i = 0; i < 16; ++i) { run += chg[base + i]; ls[i] = run; }
    sbuf[tid] = run;
    __syncthreads();
    for (int off = 1; off < 256; off <<= 1) {
        int add = (tid >= off) ? sbuf[tid - off] : 0;
        __syncthreads();
        sbuf[tid] += add;
        __syncthreads();
    }
    const int prefix = (tid > 0) ? sbuf[tid - 1] : 0;

    int si[16];
#pragma unroll
    for (int i = 0; i < 16; ++i) {
        si[i] = prefix + ls[i];
        out_seg[base + i] = si[i];
    }
#pragma unroll
    for (int i = 0; i < 16; ++i) {
        atomicAdd(&segc[si[i]], 1.0f);
        atomicAdd(&segi[si[i]], inten[base + i]);
        atomicAdd(&segf[si[i]], confp[base + i]);
    }
    __syncthreads();
#pragma unroll
    for (int i = 0; i < 16; ++i) {
        float d = fmaxf(segc[si[i]], 1.0f);
        out_mi[base + i] = segi[si[i]] / d;
        out_mc[base + i] = segf[si[i]] / d;
    }
}

// ---------------- launch ----------------
extern "C" void kernel_launch(void* const* d_in, const int* in_sizes, int n_in,
                              void* d_out, int out_size, void* d_ws, size_t ws_size,
                              hipStream_t stream)
{
    const float* gru   = (const float*)d_in[0];
    const float* inten = (const float*)d_in[1];
    const float* W1 = (const float*)d_in[2];
    const float* b1 = (const float*)d_in[3];
    const float* W2 = (const float*)d_in[4];
    const float* b2 = (const float*)d_in[5];
    const float* W3 = (const float*)d_in[6];
    const float* b3 = (const float*)d_in[7];
    const float* cw = (const float*)d_in[8];
    const float* cb = (const float*)d_in[9];

    char* ws = (char*)d_ws;
    _Float16* W1h = (_Float16*)(ws);                  // 65536 B
    _Float16* W2h = (_Float16*)(ws + 65536);          // 16384 B
    _Float16* W3h = (_Float16*)(ws + 81920);          //  2048 B
    _Float16* CWh = (_Float16*)(ws + 83968);          //  8192 B
    float* d0p  = (float*)(ws + 98304);
    float* d1p  = (float*)(ws + 98304 + 1 * 524288);
    float* d2p  = (float*)(ws + 98304 + 2 * 524288);
    int*   domp = (int*)  (ws + 98304 + 3 * 524288);
    float* confp= (float*)(ws + 98304 + 4 * 524288);
    int*   chg  = (int*)  (ws + 98304 + 5 * 524288);

    float* out    = (float*)d_out;
    float* out_pp = out;                    // (B,T,4)
    float* out_ts = out + 524288;           // (B,T)
    int*   out_sg = (int*)(out + 655360);   // (B,T) int32
    float* out_mi = out + 786432;           // (B,T)
    float* out_mc = out + 917504;           // (B,T)

    cpd_prep<<<(46080 + 255) / 256, 256, 0, stream>>>(W1, W2, W3, cw, W1h, W2h, W3h, CWh);
    cpd_mlp_wmma<<<NROWS / 64, 128, 0, stream>>>(gru, W1h, W2h, W3h, CWh, b1, b2, b3,
                                                 out_pp, d0p, d1p, d2p, domp, confp);
    cpd_trans<<<(NROWS + 255) / 256, 256, 0, stream>>>(d0p, d1p, d2p, domp, cb, out_ts, chg);
    cpd_seg<<<BZ, 256, 0, stream>>>(chg, confp, inten, out_sg, out_mi, out_mc);
}
